// ACGMultiHeadAttention_36996848287827
// MI455X (gfx1250) — compile-verified, matrix operand order checked by reading, not run on hardware
//
#include <hip/hip_runtime.h>
#include <hip/hip_bf16.h>

// ---------------------------------------------------------------------------
// ACGMultiHeadAttention for MI455X (gfx1250): bf16 WMMA everywhere.
//   B=64 S=200 H=A=256 NH=4 DH=DA=64 F=2
// Pipeline:
//   conv_act / conv_w / bias_pack  : fp32 -> bf16 staging into d_ws
//   gemm_proj (29 GEMMs)           : all linear projections, bf16 out
//   attn_kernel (256 blocks=b*h)   : fused scores+fusion+softmax+ctx
//   gemm_out (3 GEMMs)             : dense/dattr + residual + LayerNorm -> f32
// Workspace layout (bytes): act 39.3MB | wts 3.1MB | proj 190MB | ctx 19.7MB
//                           | bias table 32KB   (total ~252MB)
// ---------------------------------------------------------------------------

#define DEVFN static __device__ __forceinline__

typedef __bf16 bh_t;
typedef bh_t  bhalf16 __attribute__((ext_vector_type(16)));
typedef float floatx8 __attribute__((ext_vector_type(8)));

static constexpr int    Bn = 64, Sn = 200;
static constexpr size_t N1 = (size_t)Bn * Sn * 256;   // 3,276,800 elems per [B*S,256] matrix
static constexpr int    WELEM = 256 * 256;            // 65536

struct US4 { unsigned short x, y, z, w; };

DEVFN unsigned short f2bf(float f) {             // RNE fp32 -> bf16 bits
  unsigned int u = __float_as_uint(f);
  u += 0x7FFFu + ((u >> 16) & 1u);
  return (unsigned short)(u >> 16);
}

DEVFN floatx8 wmma_bf16(bhalf16 a, bhalf16 b, floatx8 c) {
  return __builtin_amdgcn_wmma_f32_16x16x32_bf16(false, a, false, b, (short)0, c, false, false);
}

// Load a 16x32 bf16 A/B fragment per CDNA5 layout:
//   lane m=lane&15 -> matrix row (row0+m, clamped), halves 0..7 -> K=col0+sub..+7,
//   halves 8..15 -> K=col0+16+sub..+23, sub = (lane&16)?8:0.
DEVFN bhalf16 frag_load(const unsigned short* base, int row0, int rowMax, int ld, int col0) {
  int lane = threadIdx.x & 31;
  int r = row0 + (lane & 15);
  r = (r > rowMax) ? rowMax : r;
  int sub = (lane & 16) ? 8 : 0;
  const unsigned short* p = base + (size_t)r * ld + col0 + sub;
  union { uint4 u[2]; bhalf16 v; } U;
  U.u[0] = *reinterpret_cast<const uint4*>(p);
  U.u[1] = *reinterpret_cast<const uint4*>(p + 16);
  return U.v;
}

// ------------------------------ conversion ---------------------------------
struct ConvSrc { const float* s0; const float* s1; const float* s2; const float* s3; };

__global__ __launch_bounds__(256) void conv_act(ConvSrc sp, unsigned short* act) {
  int slab = blockIdx.y;                // 0 input, 1 pos, 2/3 attr0/1, 4/5 hid0/1
  const float* src;
  switch (slab) {
    case 0:  src = sp.s0;      break;
    case 1:  src = sp.s1;      break;
    case 2:  src = sp.s2;      break;
    case 3:  src = sp.s2 + N1; break;
    case 4:  src = sp.s3;      break;
    default: src = sp.s3 + N1; break;
  }
  size_t i = ((size_t)blockIdx.x * 256 + threadIdx.x) * 4;
  float4 f = *reinterpret_cast<const float4*>(src + i);
  US4 o; o.x = f2bf(f.x); o.y = f2bf(f.y); o.z = f2bf(f.z); o.w = f2bf(f.w);
  *reinterpret_cast<US4*>(act + (size_t)slab * N1 + i) = o;
}

struct WSrc { const float* w[15]; }; // q k v qp kp dense qia qai qpa qap qL kL vL dattr qxy

__global__ __launch_bounds__(256) void conv_w(WSrc ws, unsigned short* wts) {
  const int sel[24] = {0,1,2,3,4,5, 6,6, 7,7, 8,8, 9,9, 10,10, 11,11, 12,12, 13,13, 14,14};
  const int off[24] = {0,0,0,0,0,0, 0,1, 0,1, 0,1, 0,1, 0,1,  0,1,  0,1,  0,1,  1,2};
  int wi = blockIdx.y;
  const float* src = ws.w[sel[wi]] + (size_t)off[wi] * WELEM;
  size_t i = ((size_t)blockIdx.x * 256 + threadIdx.x) * 4;
  float4 f = *reinterpret_cast<const float4*>(src + i);
  US4 o; o.x = f2bf(f.x); o.y = f2bf(f.y); o.z = f2bf(f.z); o.w = f2bf(f.w);
  *reinterpret_cast<US4*>(wts + (size_t)wi * WELEM + i) = o;
}

struct BSrc { const float* b[15]; }; // q k v qp kp qia qai qpa qap qL kL vL qxy dense dattr

__global__ __launch_bounds__(256) void bias_pack(BSrc s, float* tab) {
  const int sel[32] = {0,1,2,3,4, 9,9,10,10, 9,9,10,10, 11,11, 5,5, 6,6, 8,8,8,8, 7,7, 12,12,12,12, 13,14,14};
  const int off[32] = {0,0,0,0,0, 0,1,0,1,   0,1,0,1,   0,1,   0,1, 0,1, 0,1,0,1, 0,1, 1,2,1,2,     0,0,1};
  int t = threadIdx.x;
  for (int r = 0; r < 32; r++)
    tab[r * 256 + t] = s.b[sel[r]][off[r] * 256 + t];
}

// --------------------------- projection GEMMs ------------------------------
// Y[g] (bf16) = act[x] @ W[w]^T + bias[g] ; M=12800 N=K=256.
__global__ __launch_bounds__(256) void gemm_proj(const unsigned short* __restrict__ act,
                                                 const unsigned short* __restrict__ wts,
                                                 const float* __restrict__ biasTab,
                                                 unsigned short* __restrict__ proj) {
  const int xid[29] = {0,0,0,1,1, 2,3,2,3, 4,5,4,5, 4,5, 0,0, 2,3, 2,3, 4,5, 1,1, 2,3,4,5};
  const int wid[29] = {0,1,2,3,4, 14,15,16,17, 14,15,16,17, 18,19, 6,7, 8,9, 12,13, 12,13, 10,11, 22,23,22,23};
  extern __shared__ unsigned char smem1[];
  unsigned short* Wl = (unsigned short*)smem1;            // 128KB
  float*          bl = (float*)(smem1 + (size_t)WELEM * 2);
  int g = blockIdx.y;
  const unsigned short* X = act + (size_t)xid[g] * N1;
  const unsigned short* W = wts + (size_t)wid[g] * WELEM;
  unsigned short*       Y = proj + (size_t)g * N1;

  { const uint4* src = (const uint4*)W; uint4* dst = (uint4*)Wl;
    for (int i = threadIdx.x; i < WELEM / 8; i += 256) dst[i] = src[i]; }
  bl[threadIdx.x] = biasTab[g * 256 + threadIdx.x];
  __syncthreads();

  int wave = threadIdx.x >> 5, lane = threadIdx.x & 31;
  int cl = lane & 15, half = lane >> 4;
  int m0 = blockIdx.x * 128 + wave * 16;
  floatx8 acc[16];
#pragma unroll
  for (int nt = 0; nt < 16; nt++) acc[nt] = floatx8{0,0,0,0,0,0,0,0};

  for (int ks = 0; ks < 8; ks++) {
    bhalf16 a = frag_load(X, m0, 12799, 256, ks * 32);
#pragma unroll
    for (int nt = 0; nt < 16; nt++) {
      bhalf16 bfr = frag_load(Wl, nt * 16, 255, 256, ks * 32);
      acc[nt] = wmma_bf16(a, bfr, acc[nt]);
    }
  }
#pragma unroll
  for (int nt = 0; nt < 16; nt++) {
    int col = nt * 16 + cl;
    float bb = bl[col];
#pragma unroll
    for (int v = 0; v < 8; v++) {
      int row = m0 + v + 8 * half;
      Y[(size_t)row * 256 + col] = f2bf(acc[nt][v] + bb);
    }
  }
}

// ------------------------------ fused attention ----------------------------
// One block per (b,h). Wave w owns query row-tiles rt = w, w+8 (13 tiles of 16,
// 208 padded rows). Scores held in 13x v8f accumulators, products grouped by
// shared K matrix; weighted f32 accumulate; softmax via shfl_xor over the
// 16-lane half owning each row; probs -> per-wave LDS panel (bf16, ld 224);
// ctx = probs @ V via WMMA against LDS-staged V^T (zero padded to 224 K rows).
template <int PHASE>
DEVFN void attn_phase(const unsigned short* __restrict__ proj,
                      unsigned short* __restrict__ ctxout,
                      const float* __restrict__ mask,
                      const float* __restrict__ fusionW,
                      const float* __restrict__ fusionWA,
                      const unsigned short* VT, unsigned short* myPanel,
                      int b, int h) {
  constexpr int NG = (PHASE == 0) ? 4 : 3;   // product groups (shared K)
  constexpr int NP = (PHASE == 0) ? 4 : 3;   // products per group
  constexpr int NV = (PHASE == 0) ? 1 : 2;   // V matrices for ctx
  // projection ids: see gemm_proj table.  fused (phase0) / fused_a (phase1)
  constexpr int Kid0[4]    = {1, 4, 7, 8};
  constexpr int Kid1[4]    = {4, 11, 12, 0};
  constexpr int Qid0[4][4] = {{0,17,18,3},{0,19,20,3},{26,15,5,23},{25,16,6,24}};
  constexpr int Qid1[4][4] = {{21,22,3,0},{28,9,23,0},{27,10,24,0},{0,0,0,0}};
  constexpr int Nid0[4][4] = {{2,6,7,12},{5,10,11,15},{1,3,8,13},{0,4,9,14}};
  constexpr int Nid1[4][4] = {{4,5,8,0},{1,2,6,0},{0,3,7,0},{0,0,0,0}};

  int tid = threadIdx.x, wave = tid >> 5, lane = tid & 31;
  int cl = lane & 15, half = lane >> 4;
  int qmax = b * 200 + 199;

  for (int rt = wave; rt < 13; rt += 8) {
    floatx8 acc[13];
#pragma unroll
    for (int ct = 0; ct < 13; ct++) acc[ct] = floatx8{0,0,0,0,0,0,0,0};
    int qrow0 = b * 200 + rt * 16;

    for (int gidx = 0; gidx < NG; gidx++) {
      int kid = (PHASE == 0) ? Kid0[gidx] : Kid1[gidx];
      const unsigned short* K = proj + (size_t)kid * N1;
      bhalf16 aF[NP][2];
      float   wv[NP];
#pragma unroll
      for (int p = 0; p < NP; p++) {
        int qid = (PHASE == 0) ? Qid0[gidx][p] : Qid1[gidx][p];
        const unsigned short* Q = proj + (size_t)qid * N1;
        aF[p][0] = frag_load(Q, qrow0, qmax, 256, h * 64);
        aF[p][1] = frag_load(Q, qrow0, qmax, 256, h * 64 + 32);
        int n = (PHASE == 0) ? Nid0[gidx][p] : Nid1[gidx][p];
        wv[p] = (PHASE == 0) ? fusionW[n] : fusionWA[b * 9 + n];
      }
#pragma unroll
      for (int ct = 0; ct < 13; ct++) {
        int krow0 = b * 200 + ct * 16;
        bhalf16 b0 = frag_load(K, krow0, qmax, 256, h * 64);
        bhalf16 b1 = frag_load(K, krow0, qmax, 256, h * 64 + 32);
#pragma unroll
        for (int p = 0; p < NP; p++) {
          floatx8 t = wmma_bf16(aF[p][0], b0, floatx8{0,0,0,0,0,0,0,0});
          t = wmma_bf16(aF[p][1], b1, t);
#pragma unroll
          for (int e = 0; e < 8; e++) acc[ct][e] += wv[p] * t[e];
        }
      }
    }

    // scale + mask, then row softmax (row = (v, 16-lane half))
#pragma unroll
    for (int ct = 0; ct < 13; ct++) {
      int c = ct * 16 + cl;
#pragma unroll
      for (int v = 0; v < 8; v++) {
        int q = rt * 16 + v + 8 * half;
        float x = acc[ct][v] * 0.125f;
        if (c < 200) {
          int qq = (q < 200) ? q : 199;
          x += mask[(size_t)b * 40000 + qq * 200 + c];
        } else {
          x = -1e30f;
        }
        acc[ct][v] = x;
      }
    }
#pragma unroll
    for (int v = 0; v < 8; v++) {
      float m = -1e30f;
#pragma unroll
      for (int ct = 0; ct < 13; ct++) m = fmaxf(m, acc[ct][v]);
      for (int d = 1; d < 16; d <<= 1) m = fmaxf(m, __shfl_xor(m, d, 32));
      float s = 0.f;
#pragma unroll
      for (int ct = 0; ct < 13; ct++) {
        float e = __expf(acc[ct][v] - m);
        acc[ct][v] = e; s += e;
      }
      for (int d = 1; d < 16; d <<= 1) s += __shfl_xor(s, d, 32);
      float ri = 1.0f / s;
      int prow = v + 8 * half;
#pragma unroll
      for (int ct = 0; ct < 13; ct++)
        myPanel[prow * 224 + ct * 16 + cl] = f2bf(acc[ct][v] * ri);
    }

    // ctx = probs @ V  (K padded to 224; cols 200..223 zero)
#pragma unroll
    for (int vb = 0; vb < NV; vb++) {
      const unsigned short* vt = VT + vb * 64 * 224;
      floatx8 cacc[4];
#pragma unroll
      for (int nt = 0; nt < 4; nt++) cacc[nt] = floatx8{0,0,0,0,0,0,0,0};
      for (int ks = 0; ks < 7; ks++) {
        bhalf16 ap = frag_load(myPanel, 0, 15, 224, ks * 32);
#pragma unroll
        for (int nt = 0; nt < 4; nt++) {
          bhalf16 bf = frag_load(vt, nt * 16, 63, 224, ks * 32);
          cacc[nt] = wmma_bf16(ap, bf, cacc[nt]);
        }
      }
      unsigned short* dst = ctxout + (size_t)((PHASE == 0) ? 0 : (1 + vb)) * N1;
#pragma unroll
      for (int nt = 0; nt < 4; nt++) {
        int col = h * 64 + nt * 16 + cl;
#pragma unroll
        for (int v = 0; v < 8; v++) {
          int q = rt * 16 + v + 8 * half;
          if (q < 200)
            dst[(size_t)(b * 200 + q) * 256 + col] = f2bf(cacc[nt][v]);
        }
      }
    }
  }
}

__global__ __launch_bounds__(256) void attn_kernel(const unsigned short* __restrict__ proj,
                                                   unsigned short* __restrict__ ctxout,
                                                   const float* __restrict__ mask,
                                                   const float* __restrict__ fusionW,
                                                   const float* __restrict__ fusionWA) {
  extern __shared__ unsigned short smem2[];
  unsigned short* VT     = smem2;                 // 2 x [64][224] bf16 (V^T, zero-padded)
  unsigned short* panels = smem2 + 2 * 64 * 224;  // 8 x [16][224] bf16 probs panels
  int b = blockIdx.x >> 2, h = blockIdx.x & 3;
  int tid = threadIdx.x, wave = tid >> 5, lane = tid & 31;
  unsigned short* myPanel = panels + wave * 16 * 224;

  // zero padding columns 208..223 of this wave's panel (stay zero both phases)
  for (int i = lane; i < 256; i += 32)
    myPanel[(i >> 4) * 224 + 208 + (i & 15)] = 0;

  // ---- phase 0: fused scores with itemV ----
  for (int e = tid; e < 64 * 224; e += 256) {
    int d = e / 224, kp = e - d * 224;
    unsigned short val = 0;
    if (kp < 200) val = proj[2 * N1 + (size_t)(b * 200 + kp) * 256 + h * 64 + d];
    VT[e] = val;
  }
  __syncthreads();
  attn_phase<0>(proj, ctxout, mask, fusionW, fusionWA, VT, myPanel, b, h);
  __syncthreads();

  // ---- phase 1: fused_a scores with hidV0/hidV1 ----
  for (int vb = 0; vb < 2; vb++) {
    const unsigned short* Vsrc = proj + (size_t)(13 + vb) * N1;
    for (int e = tid; e < 64 * 224; e += 256) {
      int d = e / 224, kp = e - d * 224;
      unsigned short val = 0;
      if (kp < 200) val = Vsrc[(size_t)(b * 200 + kp) * 256 + h * 64 + d];
      VT[vb * 64 * 224 + e] = val;
    }
  }
  __syncthreads();
  attn_phase<1>(proj, ctxout, mask, fusionW, fusionWA, VT, myPanel, b, h);
}

// --------------------- output GEMM + residual + LayerNorm ------------------
struct S3P {
  const float* res0;  // input_tensor
  const float* hid;   // hidden_state_attr base
  const float* ln_g; const float* ln_b;
  const float* lnA_g; const float* lnA_b;
};

__global__ __launch_bounds__(256) void gemm_out(const unsigned short* __restrict__ ctx,
                                                const unsigned short* __restrict__ wts,
                                                const float* __restrict__ biasTab,
                                                S3P pp, float* __restrict__ out) {
  const int widG[3] = {5, 20, 21};  // W_dense, W_dattr0, W_dattr1
  extern __shared__ unsigned char smem3[];
  unsigned short* Wl = (unsigned short*)smem3;
  float* bl   = (float*)(smem3 + (size_t)WELEM * 2);
  float* gl   = bl + 256;
  float* betl = bl + 512;
  int g = blockIdx.y;
  const unsigned short* X = ctx + (size_t)g * N1;
  const unsigned short* W = wts + (size_t)widG[g] * WELEM;
  const float* res = (g == 0) ? pp.res0 : (pp.hid + (size_t)(g - 1) * N1);
  const float* gam = (g == 0) ? pp.ln_g : (pp.lnA_g + (g - 1) * 256);
  const float* bet = (g == 0) ? pp.ln_b : (pp.lnA_b + (g - 1) * 256);

  { const uint4* src = (const uint4*)W; uint4* dst = (uint4*)Wl;
    for (int i = threadIdx.x; i < WELEM / 8; i += 256) dst[i] = src[i]; }
  bl[threadIdx.x]   = biasTab[(29 + g) * 256 + threadIdx.x];
  gl[threadIdx.x]   = gam[threadIdx.x];
  betl[threadIdx.x] = bet[threadIdx.x];
  __syncthreads();

  int wave = threadIdx.x >> 5, lane = threadIdx.x & 31;
  int cl = lane & 15, half = lane >> 4;
  int m0 = blockIdx.x * 128 + wave * 16;
  floatx8 acc[16];
#pragma unroll
  for (int nt = 0; nt < 16; nt++) acc[nt] = floatx8{0,0,0,0,0,0,0,0};

  for (int ks = 0; ks < 8; ks++) {
    bhalf16 a = frag_load(X, m0, 12799, 256, ks * 32);
#pragma unroll
    for (int nt = 0; nt < 16; nt++) {
      bhalf16 bfr = frag_load(Wl, nt * 16, 255, 256, ks * 32);
      acc[nt] = wmma_bf16(a, bfr, acc[nt]);
    }
  }
  // + bias + residual
#pragma unroll
  for (int nt = 0; nt < 16; nt++) {
    int col = nt * 16 + cl;
    float bb = bl[col];
#pragma unroll
    for (int v = 0; v < 8; v++) {
      int row = m0 + v + 8 * half;
      acc[nt][v] += bb + res[(size_t)row * 256 + col];
    }
  }
  // LayerNorm per row (row fully owned by one 16-lane half across 16 tiles)
#pragma unroll
  for (int v = 0; v < 8; v++) {
    float s = 0.f, s2 = 0.f;
#pragma unroll
    for (int nt = 0; nt < 16; nt++) { float x = acc[nt][v]; s += x; s2 += x * x; }
    for (int d = 1; d < 16; d <<= 1) { s += __shfl_xor(s, d, 32); s2 += __shfl_xor(s2, d, 32); }
    float mean = s * (1.0f / 256.0f);
    float var  = s2 * (1.0f / 256.0f) - mean * mean;
    float rs   = rsqrtf(var + 1e-12f);
    int row = m0 + v + 8 * half;
#pragma unroll
    for (int nt = 0; nt < 16; nt++) {
      int col = nt * 16 + cl;
      out[(size_t)g * N1 + (size_t)row * 256 + col] =
          (acc[nt][v] - mean) * rs * gl[col] + betl[col];
    }
  }
}

// ------------------------------- launcher ----------------------------------
extern "C" void kernel_launch(void* const* d_in, const int* in_sizes, int n_in,
                              void* d_out, int out_size, void* d_ws, size_t ws_size,
                              hipStream_t stream) {
  (void)in_sizes; (void)n_in; (void)out_size; (void)ws_size;
  const float* input = (const float*)d_in[0];
  const float* attrT = (const float*)d_in[1];
  const float* pos   = (const float*)d_in[2];
  const float* hidT  = (const float*)d_in[3];
  const float* mask  = (const float*)d_in[4];
  const float* fw    = (const float*)d_in[5];
  const float* fwa   = (const float*)d_in[6];
  // params (setup_inputs dict order): W_q b_q W_k b_k W_v b_v W_qp b_qp W_kp
  // b_kp W_vp b_vp W_dense b_dense | 8x (W_*,b_*) F-indexed | W_qxy b_qxy |
  // ln_g ln_b lnA_g lnA_b
  const float* W_q = (const float*)d_in[7];   const float* b_q = (const float*)d_in[8];
  const float* W_k = (const float*)d_in[9];   const float* b_k = (const float*)d_in[10];
  const float* W_v = (const float*)d_in[11];  const float* b_v = (const float*)d_in[12];
  const float* W_qp = (const float*)d_in[13]; const float* b_qp = (const float*)d_in[14];
  const float* W_kp = (const float*)d_in[15]; const float* b_kp = (const float*)d_in[16];
  const float* W_dense = (const float*)d_in[19]; const float* b_dense = (const float*)d_in[20];
  const float* W_qia = (const float*)d_in[21]; const float* b_qia = (const float*)d_in[22];
  const float* W_qai = (const float*)d_in[23]; const float* b_qai = (const float*)d_in[24];
  const float* W_qpa = (const float*)d_in[25]; const float* b_qpa = (const float*)d_in[26];
  const float* W_qap = (const float*)d_in[27]; const float* b_qap = (const float*)d_in[28];
  const float* W_qL  = (const float*)d_in[29]; const float* b_qL  = (const float*)d_in[30];
  const float* W_kL  = (const float*)d_in[31]; const float* b_kL  = (const float*)d_in[32];
  const float* W_vL  = (const float*)d_in[33]; const float* b_vL  = (const float*)d_in[34];
  const float* W_dattr = (const float*)d_in[35]; const float* b_dattr = (const float*)d_in[36];
  const float* W_qxy = (const float*)d_in[37]; const float* b_qxy = (const float*)d_in[38];
  const float* ln_g = (const float*)d_in[39];  const float* ln_b = (const float*)d_in[40];
  const float* lnA_g = (const float*)d_in[41]; const float* lnA_b = (const float*)d_in[42];

  unsigned char* wsb = (unsigned char*)d_ws;
  size_t o = 0;
  unsigned short* act  = (unsigned short*)(wsb + o); o += 6  * N1 * 2;
  unsigned short* wts  = (unsigned short*)(wsb + o); o += 24 * (size_t)WELEM * 2;
  unsigned short* proj = (unsigned short*)(wsb + o); o += 29 * N1 * 2;
  unsigned short* ctx  = (unsigned short*)(wsb + o); o += 3  * N1 * 2;
  float* biasTab       = (float*)(wsb + o);

  ConvSrc cs{input, pos, attrT, hidT};
  conv_act<<<dim3((unsigned)(N1 / 1024), 6), 256, 0, stream>>>(cs, act);

  WSrc wsrc;
  wsrc.w[0] = W_q;  wsrc.w[1] = W_k;  wsrc.w[2] = W_v;   wsrc.w[3] = W_qp;
  wsrc.w[4] = W_kp; wsrc.w[5] = W_dense; wsrc.w[6] = W_qia; wsrc.w[7] = W_qai;
  wsrc.w[8] = W_qpa; wsrc.w[9] = W_qap; wsrc.w[10] = W_qL; wsrc.w[11] = W_kL;
  wsrc.w[12] = W_vL; wsrc.w[13] = W_dattr; wsrc.w[14] = W_qxy;
  conv_w<<<dim3(WELEM / 1024, 24), 256, 0, stream>>>(wsrc, wts);

  BSrc bsrc;
  bsrc.b[0] = b_q;  bsrc.b[1] = b_k;  bsrc.b[2] = b_v;   bsrc.b[3] = b_qp;
  bsrc.b[4] = b_kp; bsrc.b[5] = b_qia; bsrc.b[6] = b_qai; bsrc.b[7] = b_qpa;
  bsrc.b[8] = b_qap; bsrc.b[9] = b_qL; bsrc.b[10] = b_kL; bsrc.b[11] = b_vL;
  bsrc.b[12] = b_qxy; bsrc.b[13] = b_dense; bsrc.b[14] = b_dattr;
  bias_pack<<<1, 256, 0, stream>>>(bsrc, biasTab);

  size_t lds1 = (size_t)WELEM * 2 + 256 * 4;              // 132,096 B
  gemm_proj<<<dim3(100, 29), 256, lds1, stream>>>(act, wts, biasTab, proj);

  size_t lds2 = (size_t)(2 * 64 * 224 + 8 * 16 * 224) * 2; // 114,688 B
  attn_kernel<<<dim3(256), 256, lds2, stream>>>(proj, ctx, mask, fw, fwa);

  S3P s3{input, hidT, ln_g, ln_b, lnA_g, lnA_b};
  size_t lds3 = (size_t)WELEM * 2 + 3 * 256 * 4;          // 134,144 B
  gemm_out<<<dim3(100, 3), 256, lds3, stream>>>(ctx, wts, biasTab, s3, (float*)d_out);
}
